// DynamicMoELayer_69561290326695
// MI455X (gfx1250) — matrix operand order repaired and checked by reading, not compiled, not run on hardware
//
#include <hip/hip_runtime.h>
#include <hip/hip_bf16.h>
#include <math.h>

typedef __bf16 bf16_t;
typedef __attribute__((ext_vector_type(16))) __bf16 v16bf;
typedef __attribute__((ext_vector_type(8)))  __bf16 v8bf;
typedef __attribute__((ext_vector_type(8)))  float  v8f;
typedef __attribute__((ext_vector_type(4)))  unsigned int v4u;
typedef __attribute__((ext_vector_type(8)))  int v8i;
typedef __attribute__((ext_vector_type(4)))  int v4i;

#define T_TOK 8192
#define C_DIM 512
#define I_DIM 2048
#define E_NUM 16
#define EPSV  1e-12f

// ---- GEMM tiling ----
#define BM     128            // tokens per block (8 waves x 16 rows)
#define ICHUNK 64             // intermediate columns per chunk
#define LDX  (C_DIM + 8)      // 520 bf16 per row (16B-aligned, bank-spread)
#define LDW1 (C_DIM + 8)      // W1s[il][c]
#define LDW2 (ICHUNK + 8)     // W2s[c][il]
#define LDH  (ICHUNK + 8)     // Hs[m][il]

#define XS_ELEMS   (BM * LDX)
#define W1S_ELEMS  (ICHUNK * LDW1)
#define W2S_ELEMS  (C_DIM * LDW2)
#define HS_ELEMS   (BM * LDH)
static constexpr size_t SMEM_BYTES =
    (size_t)(XS_ELEMS + W1S_ELEMS + W2S_ELEMS + HS_ELEMS) * sizeof(bf16_t);
static_assert(SMEM_BYTES <= 320u * 1024u, "LDS budget exceeded");

#if defined(__gfx1250__) && __has_builtin(__builtin_amdgcn_tensor_load_to_lds) && \
    __has_builtin(__builtin_amdgcn_s_wait_tensorcnt)
#define HAVE_TDM 1
#else
#define HAVE_TDM 0
#endif

__device__ __forceinline__ v16bf make_frag(const bf16_t* p0, const bf16_t* p1) {
  v8bf lo = *reinterpret_cast<const v8bf*>(p0);
  v8bf hi = *reinterpret_cast<const v8bf*>(p1);
  v16bf r;
#pragma unroll
  for (int i = 0; i < 8; ++i) { r[i] = lo[i]; r[i + 8] = hi[i]; }
  return r;
}

__device__ __forceinline__ v8f wmma_bf16(v16bf a, v16bf b, v8f c) {
  return __builtin_amdgcn_wmma_f32_16x16x32_bf16(false, a, false, b,
                                                 (short)0, c, false, false);
}

// -------------------- f32 -> bf16 cast --------------------
__global__ void __launch_bounds__(256)
cast_bf16_kernel(const float* __restrict__ src, bf16_t* __restrict__ dst, long n) {
  long i = (long)blockIdx.x * blockDim.x + threadIdx.x;
  long stride = (long)gridDim.x * blockDim.x;
  for (; i < n; i += stride) dst[i] = (bf16_t)src[i];
}

// -------------------- routing --------------------
__global__ void __launch_bounds__(128)
routing_kernel(const float* __restrict__ x, const float* __restrict__ sim,
               const float* __restrict__ gates,
               float* __restrict__ out_logits, float* __restrict__ out_mask,
               float* __restrict__ rw) {
  __shared__ float snorm[E_NUM];
  __shared__ float thr[E_NUM];
  const int tid = threadIdx.x;
  if (tid < E_NUM) {
    float s = 0.0f;
    for (int c = 0; c < C_DIM; ++c) {
      float v = sim[c * E_NUM + tid];
      s += v * v;
    }
    snorm[tid] = fmaxf(sqrtf(s), EPSV);
    thr[tid] = 1.0f / (1.0f + expf(-gates[tid]));
  }
  __syncthreads();

  const int wave = tid >> 5;
  const int lane = tid & 31;
  const int t = blockIdx.x * 4 + wave;

  float dots[E_NUM];
#pragma unroll
  for (int e = 0; e < E_NUM; ++e) dots[e] = 0.0f;
  float ss = 0.0f;
  for (int c = lane; c < C_DIM; c += 32) {
    float xv = x[(size_t)t * C_DIM + c];
    ss += xv * xv;
    const float4* sr = reinterpret_cast<const float4*>(&sim[c * E_NUM]);
#pragma unroll
    for (int q = 0; q < 4; ++q) {
      float4 s4 = sr[q];
      dots[q * 4 + 0] += xv * s4.x;
      dots[q * 4 + 1] += xv * s4.y;
      dots[q * 4 + 2] += xv * s4.z;
      dots[q * 4 + 3] += xv * s4.w;
    }
  }
#pragma unroll
  for (int off = 16; off > 0; off >>= 1) {
    ss += __shfl_xor(ss, off, 32);
#pragma unroll
    for (int e = 0; e < E_NUM; ++e) dots[e] += __shfl_xor(dots[e], off, 32);
  }
  const float xn = fmaxf(sqrtf(ss), EPSV);

  float logit[E_NUM], gated[E_NUM], mask[E_NUM];
  float best = -1e30f;
  int bestIdx = 0, any = 0;
#pragma unroll
  for (int e = 0; e < E_NUM; ++e) {
    logit[e] = dots[e] / (xn * snorm[e]);
    if (logit[e] > best) { best = logit[e]; bestIdx = e; }
    gated[e] = fmaxf(logit[e] - thr[e], 0.0f);
    mask[e] = (gated[e] > 0.0f) ? 1.0f : 0.0f;
    any += (gated[e] > 0.0f) ? 1 : 0;
  }
  if (any == 0) {
#pragma unroll
    for (int e = 0; e < E_NUM; ++e) mask[e] = (e == bestIdx) ? 1.0f : 0.0f;
  }
  float gm[E_NUM];
  float mx = -1e30f;
#pragma unroll
  for (int e = 0; e < E_NUM; ++e) {
    gm[e] = (mask[e] > 0.0f) ? gated[e] : -1e9f;
    mx = fmaxf(mx, gm[e]);
  }
  float denom = 0.0f, w[E_NUM];
#pragma unroll
  for (int e = 0; e < E_NUM; ++e) { w[e] = expf(gm[e] - mx); denom += w[e]; }
  const float inv = 1.0f / denom;
  if (lane == 0) {
#pragma unroll
    for (int e = 0; e < E_NUM; ++e) {
      out_logits[(size_t)t * E_NUM + e] = logit[e];
      out_mask[(size_t)t * E_NUM + e]   = mask[e];
      rw[(size_t)t * E_NUM + e]         = w[e] * inv;
    }
  }
}

// -------------------- fused per-expert double GEMM --------------------
__global__ void __launch_bounds__(256)
moe_expert_gemm(const bf16_t* __restrict__ xbf, const bf16_t* __restrict__ w1bf,
                const bf16_t* __restrict__ w2bf, const float* __restrict__ maskv,
                float* __restrict__ feo) {
  extern __shared__ bf16_t smem[];
  bf16_t* Xs  = smem;
  bf16_t* W1s = Xs + XS_ELEMS;
  bf16_t* W2s = W1s + W1S_ELEMS;
  bf16_t* Hs  = W2s + W2S_ELEMS;

  const int tid  = threadIdx.x;
  const int wave = tid >> 5;
  const int lane = tid & 31;
  const int l    = lane & 15;
  const int sel  = lane >> 4;  // 0: k 0-7,16-23; 1: k 8-15,24-31
  const int e    = blockIdx.y;
  const int t0   = blockIdx.x * BM;
  const int m0   = wave * 16;

  const bf16_t* w1e = w1bf + (size_t)e * C_DIM * I_DIM;  // [C][I] row-major
  const bf16_t* w2e = w2bf + (size_t)e * I_DIM * C_DIM;  // [I][C] row-major

#if HAVE_TDM
  // Tensor Data Mover: DMA the X tile [BM x C_DIM] bf16 into LDS, inserting
  // 8 bf16 (4 DWORDs) of padding after every 512-bf16 row -> row pitch LDX.
  if (wave == 0) {
    const unsigned ldsoff = (unsigned)(uintptr_t)Xs;   // LDS byte offset
    const unsigned long long ga =
        (unsigned long long)(uintptr_t)(xbf + (size_t)t0 * C_DIM);
    v4u g0;
    g0[0] = 1u;                                        // count=1 (valid D#)
    g0[1] = ldsoff;                                    // lds_addr
    g0[2] = (unsigned)(ga & 0xFFFFFFFFu);              // global_addr[31:0]
    g0[3] = (unsigned)((ga >> 32) & 0x01FFFFFFu) | (2u << 30);  // addr hi|type=2
    v8i g1;
    g1[0] = (int)((1u << 16)          // data_size = 1 -> 2 bytes
                | (1u << 20)          // pad_enable
                | (7u << 22)          // pad_interval: 256 DWORDs (one row)
                | (3u << 25));        // pad_amount: 4 DWORDs (8 bf16)
    g1[1] = (int)((unsigned)(C_DIM & 0xFFFF) << 16);   // tensor_dim0 lo16
    g1[2] = (int)((unsigned)(T_TOK & 0xFFFF) << 16);   // td0 hi=0 | tensor_dim1 lo16
    g1[3] = (int)((unsigned)C_DIM << 16);              // td1 hi=0 | tile_dim0=512
    g1[4] = (int)BM;                                   // tile_dim1=128, tile_dim2=0
    g1[5] = (int)C_DIM;                                // tensor_dim0_stride = 512
    g1[6] = 0;
    g1[7] = 0;
    v4i gz4 = {0, 0, 0, 0};
    v8i gz8 = {0, 0, 0, 0, 0, 0, 0, 0};
    // 6-arg toolchain variant: (g0, g1, g2, g3, extra_group, cpol)
    __builtin_amdgcn_tensor_load_to_lds(g0, g1, gz4, gz4, gz8, 0);
    __builtin_amdgcn_s_wait_tensorcnt(0);
  }
#else
  // manual staging of X tile [BM x C] (bf16, vectorized 16B)
  for (int k = tid; k < (BM * C_DIM) / 8; k += 256) {
    int r  = k >> 6;
    int c8 = (k & 63) << 3;
    v8bf v = *reinterpret_cast<const v8bf*>(&xbf[(size_t)(t0 + r) * C_DIM + c8]);
    *reinterpret_cast<v8bf*>(&Xs[r * LDX + c8]) = v;
  }
#endif

  v8f oacc[32];
#pragma unroll
  for (int nt = 0; nt < 32; ++nt)
#pragma unroll
    for (int v = 0; v < 8; ++v) oacc[nt][v] = 0.0f;

  for (int ic = 0; ic < I_DIM / ICHUNK; ++ic) {
    const int i0 = ic * ICHUNK;
    __syncthreads();  // prior readers of W1s/W2s done (also orders X staging)

    // prefetch next weight chunk toward L2/WGP$ (global_prefetch_b8)
    if (ic + 1 < I_DIM / ICHUNK) {
      const int ip = i0 + ICHUNK;
      const bf16_t* p1 = &w1e[(size_t)(tid << 1) * I_DIM + ip];
      __builtin_prefetch(p1, 0, 1);
      __builtin_prefetch(p1 + I_DIM, 0, 1);
      const bf16_t* p2 = &w2e[(size_t)(ip + (tid >> 2)) * C_DIM + ((tid & 3) << 7)];
      __builtin_prefetch(p2, 0, 1);
    }

    // stage W1 chunk transposed: W1s[il][c] = w1[c][i0+il]
    for (int k = tid; k < (C_DIM * ICHUNK) / 8; k += 256) {
      int c   = k >> 3;
      int il8 = (k & 7) << 3;
      v8bf v = *reinterpret_cast<const v8bf*>(&w1e[(size_t)c * I_DIM + i0 + il8]);
#pragma unroll
      for (int j = 0; j < 8; ++j) W1s[(il8 + j) * LDW1 + c] = v[j];
    }
    // stage W2 chunk transposed: W2s[c][il] = w2[i0+il][c]
    for (int k = tid; k < (C_DIM * ICHUNK) / 8; k += 256) {
      int il = k >> 6;
      int c8 = (k & 63) << 3;
      v8bf v = *reinterpret_cast<const v8bf*>(&w2e[(size_t)(i0 + il) * C_DIM + c8]);
#pragma unroll
      for (int j = 0; j < 8; ++j) W2s[(c8 + j) * LDW2 + il] = v[j];
    }
    __syncthreads();

    // GEMM1: h[16 x ICHUNK] = Xtile[16 x 512] @ W1chunk[512 x ICHUNK]
    v8f hacc[ICHUNK / 16];
#pragma unroll
    for (int nt = 0; nt < ICHUNK / 16; ++nt)
#pragma unroll
      for (int v = 0; v < 8; ++v) hacc[nt][v] = 0.0f;

    for (int k0 = 0; k0 < C_DIM; k0 += 32) {
      const bf16_t* ap = &Xs[(m0 + l) * LDX + k0 + sel * 8];
      v16bf a = make_frag(ap, ap + 16);
      // issue all 4 B-fragment loads before the WMMA burst (overlap DS latency)
      v16bf b[ICHUNK / 16];
#pragma unroll
      for (int nt = 0; nt < ICHUNK / 16; ++nt) {
        const bf16_t* bp = &W1s[(nt * 16 + l) * LDW1 + k0 + sel * 8];
        b[nt] = make_frag(bp, bp + 16);
      }
#pragma unroll
      for (int nt = 0; nt < ICHUNK / 16; ++nt)
        hacc[nt] = wmma_bf16(a, b[nt], hacc[nt]);
    }

    // exact-erf GELU, store h rows (wave-private) to LDS as bf16
#pragma unroll
    for (int nt = 0; nt < ICHUNK / 16; ++nt) {
#pragma unroll
      for (int v = 0; v < 8; ++v) {
        int m = m0 + v + sel * 8;
        float hv = hacc[nt][v];
        float g = 0.5f * hv * (1.0f + erff(hv * 0.70710678118654752f));
        Hs[m * LDH + nt * 16 + l] = (bf16_t)g;
      }
    }
    // Hs rows are produced and consumed by the same wave: DS in-order, no barrier

    // GEMM2: out[16 x 512] += h[16 x ICHUNK] @ W2chunk[ICHUNK x 512]
    // B-fragments double-buffered so WMMA doesn't serialize on s_wait_dscnt 0
#pragma unroll
    for (int k0 = 0; k0 < ICHUNK; k0 += 32) {
      const bf16_t* ap = &Hs[(m0 + l) * LDH + k0 + sel * 8];
      v16bf a = make_frag(ap, ap + 16);
#pragma unroll
      for (int nt = 0; nt < 32; nt += 2) {
        const bf16_t* bp0 = &W2s[((nt + 0) * 16 + l) * LDW2 + k0 + sel * 8];
        const bf16_t* bp1 = &W2s[((nt + 1) * 16 + l) * LDW2 + k0 + sel * 8];
        v16bf b0 = make_frag(bp0, bp0 + 16);
        v16bf b1 = make_frag(bp1, bp1 + 16);
        oacc[nt + 0] = wmma_bf16(a, b0, oacc[nt + 0]);
        oacc[nt + 1] = wmma_bf16(a, b1, oacc[nt + 1]);
      }
    }
  }

  // epilogue: apply activation mask, write full_expert_outputs [T,E,C]
#pragma unroll
  for (int v = 0; v < 8; ++v) {
    const int m = m0 + v + sel * 8;
    const int t = t0 + m;
    const float mk = maskv[(size_t)t * E_NUM + e];
    float* dst = &feo[((size_t)t * E_NUM + e) * C_DIM];
#pragma unroll
    for (int nt = 0; nt < 32; ++nt) dst[nt * 16 + l] = oacc[nt][v] * mk;
  }
}

// -------------------- final weighted sum over experts --------------------
__global__ void __launch_bounds__(256)
finalize_kernel(const float* __restrict__ feo, const float* __restrict__ rw,
                float* __restrict__ outf) {
  const size_t idx = (size_t)blockIdx.x * 256 + threadIdx.x;
  const int t = (int)(idx >> 9);
  const float* rwt = &rw[(size_t)t * E_NUM];
  const float* f = &feo[((size_t)t * E_NUM) * C_DIM + (idx & 511)];
  float s = 0.0f;
#pragma unroll
  for (int e = 0; e < E_NUM; ++e) s += rwt[e] * f[(size_t)e * C_DIM];
  outf[idx] = s;
}

// -------------------- launcher --------------------
extern "C" void kernel_launch(void* const* d_in, const int* in_sizes, int n_in,
                              void* d_out, int out_size, void* d_ws, size_t ws_size,
                              hipStream_t stream) {
  (void)in_sizes; (void)n_in; (void)out_size; (void)ws_size;
  const float* x     = (const float*)d_in[0];
  const float* sim   = (const float*)d_in[1];
  const float* gates = (const float*)d_in[2];
  const float* w1    = (const float*)d_in[3];
  const float* w2    = (const float*)d_in[4];

  float* out_final  = (float*)d_out;
  float* out_feo    = out_final + (size_t)T_TOK * C_DIM;
  float* out_logits = out_feo + (size_t)T_TOK * E_NUM * C_DIM;
  float* out_mask   = out_logits + (size_t)T_TOK * E_NUM;

  char* ws = (char*)d_ws;
  float* rw = (float*)ws;
  size_t off = ((size_t)T_TOK * E_NUM * sizeof(float) + 255) & ~(size_t)255;
  bf16_t* xbf = (bf16_t*)(ws + off);
  off += ((size_t)T_TOK * C_DIM * sizeof(bf16_t) + 255) & ~(size_t)255;
  bf16_t* w1bf = (bf16_t*)(ws + off);
  off += ((size_t)E_NUM * C_DIM * I_DIM * sizeof(bf16_t) + 255) & ~(size_t)255;
  bf16_t* w2bf = (bf16_t*)(ws + off);

  cast_bf16_kernel<<<4096, 256, 0, stream>>>(x, xbf, (long)T_TOK * C_DIM);
  cast_bf16_kernel<<<8192, 256, 0, stream>>>(w1, w1bf, (long)E_NUM * C_DIM * I_DIM);
  cast_bf16_kernel<<<8192, 256, 0, stream>>>(w2, w2bf, (long)E_NUM * I_DIM * C_DIM);

  routing_kernel<<<T_TOK / 4, 128, 0, stream>>>(x, sim, gates, out_logits,
                                                out_mask, rw);

  dim3 gg(T_TOK / BM, E_NUM);
  moe_expert_gemm<<<gg, 256, SMEM_BYTES, stream>>>(xbf, w1bf, w2bf, out_mask,
                                                   out_feo);

  finalize_kernel<<<(T_TOK * C_DIM) / 256, 256, 0, stream>>>(out_feo, rw,
                                                             out_final);
}
